// Model_87840671138041
// MI455X (gfx1250) — compile-verified, matrix operand order checked by reading, not run on hardware
//
#include <hip/hip_runtime.h>

typedef __attribute__((ext_vector_type(16))) _Float16 v16h;
typedef __attribute__((ext_vector_type(8)))  float    v8f;

#define N_NODES   525080
#define B_GRAPHS  40
#define NPG       (N_NODES / B_GRAPHS)   // 13127
#define NC_ROWS   36000
#define E_EDGES   896088
#define NF        140
#define FEAT_EMB  20
#define OPC_EMB   12
#define CFG_DIM   18
#define IN_CH0    32
#define MAX_CH    256

// ---------------------------------------------------------------------------
// cfg scatter: cfg_all[ids[i] + ptr[batch[i]]] = node_config_feat[i]
// (last-write-wins; duplicate-id write order unspecified, matching jnp .set)
// ---------------------------------------------------------------------------
__global__ void scatter_cfg(const float* __restrict__ ncf, const int* __restrict__ ids,
                            const int* __restrict__ cbatch, const int* __restrict__ ptr,
                            float* __restrict__ cfg, int nc) {
  int i = blockIdx.x * blockDim.x + threadIdx.x;
  if (i >= nc) return;
  int g = ids[i] + ptr[cbatch[i]];
  const float* s = ncf + (size_t)i * CFG_DIM;
  float* d = cfg + (size_t)g * CFG_DIM;
#pragma unroll
  for (int k = 0; k < CFG_DIM; ++k) d[k] = s[k];
}

// ---------------------------------------------------------------------------
// featurize: x0 = relu( concat(log1p(relu(f))@Wfe^T + bfe, opc_emb, cfg) @ Win^T + bin )
// one thread per node; weight reads are wave-uniform -> scalar loads
// ---------------------------------------------------------------------------
__global__ void featurize(const float* __restrict__ nf, const int* __restrict__ opc,
                          const float* __restrict__ cfg,
                          const float* __restrict__ few, const float* __restrict__ feb,
                          const float* __restrict__ oemb,
                          const float* __restrict__ inw, const float* __restrict__ inb,
                          float* __restrict__ x0, int n) {
  int i = blockIdx.x * blockDim.x + threadIdx.x;
  if (i >= n) return;
  float in50[FEAT_EMB + OPC_EMB + CFG_DIM];
  float acc[FEAT_EMB];
#pragma unroll
  for (int j = 0; j < FEAT_EMB; ++j) acc[j] = feb[j];
  const float* frow = nf + (size_t)i * NF;
  for (int k = 0; k < NF; ++k) {
    float v = frow[k];
    v = v > 0.f ? v : 0.f;
    v = log1pf(v);
#pragma unroll
    for (int j = 0; j < FEAT_EMB; ++j) acc[j] += v * few[j * NF + k];
  }
#pragma unroll
  for (int j = 0; j < FEAT_EMB; ++j) in50[j] = acc[j];
  int oc = opc[i];
#pragma unroll
  for (int j = 0; j < OPC_EMB; ++j) in50[FEAT_EMB + j] = oemb[oc * OPC_EMB + j];
  const float* crow = cfg + (size_t)i * CFG_DIM;
#pragma unroll
  for (int j = 0; j < CFG_DIM; ++j) in50[FEAT_EMB + OPC_EMB + j] = crow[j];
  float* xrow = x0 + (size_t)i * IN_CH0;
  for (int o = 0; o < IN_CH0; ++o) {
    float a = inb[o];
    const float* wrow = inw + o * (FEAT_EMB + OPC_EMB + CFG_DIM);
#pragma unroll
    for (int k = 0; k < FEAT_EMB + OPC_EMB + CFG_DIM; ++k) a += in50[k] * wrow[k];
    xrow[o] = fmaxf(a, 0.f);
  }
}

// ---------------------------------------------------------------------------
// in-degree + inverse
// ---------------------------------------------------------------------------
__global__ void degree_k(const int* __restrict__ dst, float* __restrict__ deg, int e) {
  int i = blockIdx.x * blockDim.x + threadIdx.x;
  if (i >= e) return;
  atomicAdd(&deg[dst[i]], 1.0f);
}
__global__ void invdeg_k(const float* __restrict__ deg, float* __restrict__ inv, int n) {
  int i = blockIdx.x * blockDim.x + threadIdx.x;
  if (i >= n) return;
  inv[i] = 1.0f / fmaxf(deg[i], 1.0f);
}

// ---------------------------------------------------------------------------
// scatter-add: agg[dst] += x[src]; one thread per (edge, 4 channels)
// float4 gather + 4x global_atomic_add_f32
// ---------------------------------------------------------------------------
__global__ void scatter_add(const int* __restrict__ src, const int* __restrict__ dst,
                            const float* __restrict__ x, float* __restrict__ agg,
                            int e, int ch) {
  int per = ch >> 2;
  long long t = (long long)blockIdx.x * blockDim.x + threadIdx.x;
  if (t >= (long long)e * per) return;
  int ed = (int)(t / per);
  int cg = ((int)(t % per)) << 2;
  int s = src[ed], d = dst[ed];
  const float4 v = *(const float4*)(x + (size_t)s * ch + cg);
  float* p = agg + (size_t)d * ch + cg;
  atomicAdd(p + 0, v.x);
  atomicAdd(p + 1, v.y);
  atomicAdd(p + 2, v.z);
  atomicAdd(p + 3, v.w);
}

// ---------------------------------------------------------------------------
// build f16 A operand: ab[i][0:ch] = agg[i]*inv[i]; ab[i][ch:2ch] = x[i]
// ---------------------------------------------------------------------------
__global__ void build_abuf(const float* __restrict__ agg, const float* __restrict__ x,
                           const float* __restrict__ inv, _Float16* __restrict__ ab,
                           int n, int ch) {
  int K = ch << 1;
  long long t = (long long)blockIdx.x * blockDim.x + threadIdx.x;
  if (t >= (long long)n * K) return;
  int i = (int)(t / K);
  int k = (int)(t % K);
  float v = (k < ch) ? agg[(size_t)i * ch + k] * inv[i]
                     : x[(size_t)i * ch + (k - ch)];
  ab[(size_t)i * K + k] = (_Float16)v;
}

// ---------------------------------------------------------------------------
// pack combined weights B[k][n] (k<ch: lin_l_w[n][k]; else lin_r_w[n][k-ch])
// into per-lane WMMA B-fragment layout:
//   tile (nt,kt): 32 lanes x 16 halfs contiguous.
//   lane<16  -> col nt*16+lane, K k0+{0..7} then k0+{16..23}
//   lane>=16 -> same col,       K k0+{8..15} then k0+{24..31}
// ---------------------------------------------------------------------------
__global__ void pack_w(const float* __restrict__ wl, const float* __restrict__ wr,
                       _Float16* __restrict__ wp, int ch, int out) {
  int K = ch << 1;
  int total = K * out;
  int t = blockIdx.x * blockDim.x + threadIdx.x;
  if (t >= total) return;
  int j = t & 15;
  int lane = (t >> 4) & 31;
  int tile = t >> 9;
  int kTiles = K >> 5;
  int kt = tile % kTiles;
  int nt = tile / kTiles;
  int n = nt * 16 + (lane & 15);
  int k = kt * 32 + ((lane & 16) ? ((j < 8) ? 8 + j : 16 + j)
                                 : ((j < 8) ? j     : 8 + j));
  float v = (k < ch) ? wl[n * ch + k] : wr[n * ch + (k - ch)];
  wp[t] = (_Float16)v;
}

// ---------------------------------------------------------------------------
// WMMA GEMM, N-register-blocked: one wave computes a 16(M) x 64(N) tile.
// A fragment loaded once per K-step and reused across 4 WMMAs
// (2.5 b128 loads per v_wmma instead of 4). All out channels are
// multiples of 64 so the 4-tile grouping is exact.
// ---------------------------------------------------------------------------
__global__ void __launch_bounds__(256) sage_gemm(
    const _Float16* __restrict__ A, const _Float16* __restrict__ Wp,
    const float* __restrict__ bias, float* __restrict__ Xo,
    int n, int K, int out) {
  int lane = threadIdx.x & 31;
  int wv = blockIdx.x * (blockDim.x >> 5) + (threadIdx.x >> 5);
  int nG = out >> 6;                      // groups of 4 n-tiles (64 columns)
  int mT = (n + 15) >> 4;
  if (wv >= mT * nG) return;
  int mt = wv / nG;
  int ng = wv % nG;

  int rowA = mt * 16 + (lane & 15);
  if (rowA >= n) rowA = n - 1;            // clamp (partial last m-tile)
  const _Float16* arow = A + (size_t)rowA * K + ((lane & 16) ? 8 : 0);
  int kTiles = K >> 5;
  size_t ns = (size_t)kTiles * 512;       // halfs between consecutive n-tiles
  const _Float16* wbase = Wp + (size_t)(ng * 4) * ns + (size_t)lane * 16;

  v8f acc0 = {}, acc1 = {}, acc2 = {}, acc3 = {};
  for (int kt = 0; kt < kTiles; ++kt) {
    v16h a;
    *(uint4*)&a         = *(const uint4*)(arow + kt * 32);        // K lo half
    *(((uint4*)&a) + 1) = *(const uint4*)(arow + kt * 32 + 16);   // K hi half
    const _Float16* wt = wbase + (size_t)kt * 512;
    v16h b0, b1, b2, b3;
    *(uint4*)&b0         = *(const uint4*)(wt);
    *(((uint4*)&b0) + 1) = *(const uint4*)(wt + 8);
    *(uint4*)&b1         = *(const uint4*)(wt + ns);
    *(((uint4*)&b1) + 1) = *(const uint4*)(wt + ns + 8);
    *(uint4*)&b2         = *(const uint4*)(wt + 2 * ns);
    *(((uint4*)&b2) + 1) = *(const uint4*)(wt + 2 * ns + 8);
    *(uint4*)&b3         = *(const uint4*)(wt + 3 * ns);
    *(((uint4*)&b3) + 1) = *(const uint4*)(wt + 3 * ns + 8);
    acc0 = __builtin_amdgcn_wmma_f32_16x16x32_f16(false, a, false, b0, (short)0, acc0, false, false);
    acc1 = __builtin_amdgcn_wmma_f32_16x16x32_f16(false, a, false, b1, (short)0, acc1, false, false);
    acc2 = __builtin_amdgcn_wmma_f32_16x16x32_f16(false, a, false, b2, (short)0, acc2, false, false);
    acc3 = __builtin_amdgcn_wmma_f32_16x16x32_f16(false, a, false, b3, (short)0, acc3, false, false);
  }

  int rbase = mt * 16 + ((lane & 16) ? 8 : 0);
#define SAGE_EPI(ACC, G)                                                    \
  {                                                                         \
    int col = (ng * 4 + (G)) * 16 + (lane & 15);                            \
    float bv = bias[col];                                                   \
    _Pragma("unroll") for (int v = 0; v < 8; ++v) {                         \
      int r = rbase + v;                                                    \
      if (r < n) Xo[(size_t)r * out + col] = fmaxf(ACC[v] + bv, 0.f);       \
    }                                                                       \
  }
  SAGE_EPI(acc0, 0)
  SAGE_EPI(acc1, 1)
  SAGE_EPI(acc2, 2)
  SAGE_EPI(acc3, 3)
#undef SAGE_EPI
}

// ---------------------------------------------------------------------------
// readout: per-node dot(x, out_w) + out_b, summed per graph (graphs are
// contiguous NPG-node blocks). LDS partials -> few global atomics per block.
// ---------------------------------------------------------------------------
__global__ void readout(const float* __restrict__ x, const float* __restrict__ ow,
                        const float* __restrict__ ob, float* __restrict__ gsum,
                        int n, int ch) {
  __shared__ float part[B_GRAPHS];
  for (int b = threadIdx.x; b < B_GRAPHS; b += blockDim.x) part[b] = 0.f;
  __syncthreads();
  int i = blockIdx.x * blockDim.x + threadIdx.x;
  if (i < n) {
    float a = ob[0];
    const float* xr = x + (size_t)i * ch;
    for (int k = 0; k < ch; ++k) a += xr[k] * ow[k];
    atomicAdd(&part[i / NPG], a);
  }
  __syncthreads();
  for (int b = threadIdx.x; b < B_GRAPHS; b += blockDim.x)
    if (part[b] != 0.f) atomicAdd(&gsum[b], part[b]);
}

__global__ void finalize(const float* __restrict__ gsum, float* __restrict__ out) {
  int b = threadIdx.x;
  if (b < B_GRAPHS) out[b] = 0.001f * gsum[b];
}

// ---------------------------------------------------------------------------
extern "C" void kernel_launch(void* const* d_in, const int* in_sizes, int n_in,
                              void* d_out, int out_size, void* d_ws, size_t ws_size,
                              hipStream_t stream) {
  (void)in_sizes; (void)n_in; (void)out_size; (void)ws_size;
  // ---- inputs (setup_inputs() dict order; params leaves in sorted pytree order)
  const float* node_feat  = (const float*)d_in[0];
  const int*   node_opc   = (const int*)  d_in[1];
  const int*   ptr        = (const int*)  d_in[3];
  const float* ncf        = (const float*)d_in[4];
  const int*   nci        = (const int*)  d_in[5];
  const int*   ncb        = (const int*)  d_in[6];
  const int*   ei         = (const int*)  d_in[8];
  const int*   src = ei;
  const int*   dst = ei + E_EDGES;
  const float* lin_l_b[6]; const float* lin_l_w[6]; const float* lin_r_w[6];
  for (int l = 0; l < 6; ++l) {
    lin_l_b[l] = (const float*)d_in[9  + 3 * l];
    lin_l_w[l] = (const float*)d_in[10 + 3 * l];
    lin_r_w[l] = (const float*)d_in[11 + 3 * l];
  }
  const float* feb  = (const float*)d_in[27];
  const float* few  = (const float*)d_in[28];
  const float* inb  = (const float*)d_in[29];
  const float* inw  = (const float*)d_in[30];
  const float* oemb = (const float*)d_in[31];
  const float* outb = (const float*)d_in[32];
  const float* outw = (const float*)d_in[33];

  // ---- workspace partition
  const size_t NB = (size_t)N_NODES;
  char* w = (char*)d_ws;
  auto take = [&](size_t bytes) { char* p = w; w += (bytes + 255) & ~(size_t)255; return p; };
  float*    xA   = (float*)   take(NB * MAX_CH * 4);
  float*    xB   = (float*)   take(NB * MAX_CH * 4);
  float*    agg  = (float*)   take(NB * MAX_CH * 4);
  _Float16* abuf = (_Float16*)take(NB * 2 * MAX_CH * 2);
  float*    cfg  = (float*)   take(NB * CFG_DIM * 4);
  float*    deg  = (float*)   take(NB * 4);
  float*    inv  = (float*)   take(NB * 4);
  float*    gsum = (float*)   take(B_GRAPHS * 4);
  _Float16* wp[6];
  const int channels[7] = {32, 64, 64, 128, 128, 256, 256};
  for (int l = 0; l < 6; ++l)
    wp[l] = (_Float16*)take((size_t)2 * channels[l] * channels[l + 1] * 2);

  const int TB = 256;
  auto grid1 = [](long long t, int b) { return dim3((unsigned)((t + b - 1) / b)); };

  // ---- stage 0: cfg scatter + featurize + degrees + weight packs
  hipMemsetAsync(cfg,  0, NB * CFG_DIM * 4, stream);
  hipMemsetAsync(deg,  0, NB * 4,           stream);
  hipMemsetAsync(gsum, 0, B_GRAPHS * 4,     stream);
  scatter_cfg<<<grid1(NC_ROWS, TB), TB, 0, stream>>>(ncf, nci, ncb, ptr, cfg, NC_ROWS);
  featurize<<<grid1(N_NODES, TB), TB, 0, stream>>>(node_feat, node_opc, cfg,
                                                   few, feb, oemb, inw, inb, xA, N_NODES);
  degree_k<<<grid1(E_EDGES, TB), TB, 0, stream>>>(dst, deg, E_EDGES);
  invdeg_k<<<grid1(N_NODES, TB), TB, 0, stream>>>(deg, inv, N_NODES);
  for (int l = 0; l < 6; ++l) {
    int tot = 2 * channels[l] * channels[l + 1];
    pack_w<<<grid1(tot, TB), TB, 0, stream>>>(lin_l_w[l], lin_r_w[l], wp[l],
                                              channels[l], channels[l + 1]);
  }

  // ---- stage 1: 6 SAGE layers
  float* xbuf[2] = {xA, xB};
  int cur = 0;
  for (int l = 0; l < 6; ++l) {
    int in = channels[l], out = channels[l + 1], K = 2 * in;
    hipMemsetAsync(agg, 0, NB * in * 4, stream);
    scatter_add<<<grid1((long long)E_EDGES * (in >> 2), TB), TB, 0, stream>>>(
        src, dst, xbuf[cur], agg, E_EDGES, in);
    build_abuf<<<grid1((long long)N_NODES * K, TB), TB, 0, stream>>>(
        agg, xbuf[cur], inv, abuf, N_NODES, in);
    long long waves = (long long)((N_NODES + 15) / 16) * (out / 64);
    sage_gemm<<<grid1(waves * 32, TB), TB, 0, stream>>>(
        abuf, wp[l], lin_l_b[l], xbuf[1 - cur], N_NODES, K, out);
    cur = 1 - cur;
  }

  // ---- stage 2: readout
  readout<<<grid1(N_NODES, TB), TB, 0, stream>>>(xbuf[cur], outw, outb, gsum,
                                                 N_NODES, channels[6]);
  finalize<<<1, 64, 0, stream>>>(gsum, (float*)d_out);
}
